// RecommendationModel_25658134626988
// MI455X (gfx1250) — matrix operand order repaired and checked by reading, not compiled, not run on hardware
//
#include <hip/hip_runtime.h>
#include <math.h>

#define HID   256
#define NHEAD 8
#define DHEAD 32
#define NU_   20000
#define NI_   50000
#define NT_   50000
#define NIM_  50000
#define EB_   400000
#define ETA_  200000
#define EIA_  200000

typedef __attribute__((ext_vector_type(16))) __bf16 v16bf;
typedef __attribute__((ext_vector_type(8)))  float  v8f;

#define TM    128   // block rows   (8 waves x 16)
#define TN    128   // block cols   (8 WMMA col-tiles, A fragment reused 8x)
#define ASTR  34    // padded LDS stride (bf16 elems)
#define BSTR  34

// Native-transcendental GELU (tanh approx): v_exp_f32 + v_rcp_f32, no libm branches.
__device__ __forceinline__ float fast_gelu(float x)
{
    float u  = 0.7978845608028654f * (x + 0.044715f * x * x * x);
    float uc = fminf(fmaxf(u, -9.0f), 9.0f);                 // avoid inf/inf
    float e  = __builtin_amdgcn_exp2f(2.8853900817779268f * uc);  // exp(2u)
    float th = (e - 1.0f) * __builtin_amdgcn_rcpf(e + 1.0f); // tanh(u)
    return 0.5f * x * (1.0f + th);
}

__device__ __forceinline__ float fast_sigmoid(float x)
{
    float e = __builtin_amdgcn_exp2f(-1.4426950408889634f * x);
    return __builtin_amdgcn_rcpf(1.0f + e);
}

// ---------------------------------------------------------------------------
// GEMM: C[M,256] = pre(A)[M,256] @ W[256,256] + bias, bf16 WMMA f32-acc.
// Block tile 128x128: each wave holds one A fragment (16x32) and sweeps 8
// B fragments -> 8 accumulators, 64 v_wmma per wave per GEMM.
// ---------------------------------------------------------------------------
__global__ __launch_bounds__(256)
void hgt_gemm_bf16_wmma(const float* __restrict__ A, const float* __restrict__ W,
                        const float* __restrict__ bias, float* __restrict__ C,
                        int M, int preGelu,
                        const float* __restrict__ xres, const float* __restrict__ skip)
{
    __shared__ __bf16 sA[TM * ASTR];   // 128 rows x 32 k (+pad)
    __shared__ __bf16 sBT[TN * BSTR];  // B^T: 128 cols x 32 k (+pad)

    const int t    = threadIdx.x;
    const int lane = t & 31;
    const int wave = t >> 5;
    const int rowBase = blockIdx.x * TM;
    const int colBase = blockIdx.y * TN;
    const int half    = (lane >= 16) ? 1 : 0;
    const int mrow    = wave * 16 + (lane & 15);

    v8f acc[8];
    #pragma unroll
    for (int ct = 0; ct < 8; ++ct) acc[ct] = (v8f){};

    for (int k0 = 0; k0 < HID; k0 += 32) {
        // Stage A tile 128x32 (f32 -> bf16); a wave covers one 128B row chunk.
        #pragma unroll
        for (int i = 0; i < 16; ++i) {
            int flat = t + i * 256;            // 0..4095
            int r = flat >> 5, c = flat & 31;
            int gr = rowBase + r;
            float v = 0.0f;
            if (gr < M) {
                const float* ap = &A[(size_t)gr * HID + k0 + c];
                v = *ap;
                if (k0 + 32 < HID) __builtin_prefetch(ap + 32, 0, 1);  // global_prefetch_b8
            }
            if (preGelu) v = fast_gelu(v);
            sA[r * ASTR + c] = (__bf16)v;
        }
        // Stage B tile transposed: sBT[col][k] = W[k0+k][colBase+col]
        #pragma unroll
        for (int i = 0; i < 16; ++i) {
            int flat = t + i * 256;            // 0..4095
            int r = flat >> 7, c = flat & 127; // r = k (0..31), c = col (0..127)
            sBT[c * BSTR + r] = (__bf16)W[(size_t)(k0 + r) * HID + colBase + c];
        }
        __syncthreads();

        // One A fragment per wave (ISA 7.12.2 16-bit A 16x32 layout).
        union { v16bf v; unsigned u[8]; } af;
        #pragma unroll
        for (int j = 0; j < 8; ++j) {
            int kk = (j < 4 ? 2 * j : 16 + 2 * (j - 4)) + half * 8;   // even -> 4B aligned
            af.u[j] = *(const unsigned*)&sA[mrow * ASTR + kk];
        }
        // Sweep 8 B fragments, reusing A.
        #pragma unroll
        for (int ct = 0; ct < 8; ++ct) {
            union { v16bf v; unsigned u[8]; } bf;
            const int bcol = ct * 16 + (lane & 15);
            #pragma unroll
            for (int j = 0; j < 8; ++j) {
                int kb = (j < 4 ? 2 * j : 16 + 2 * (j - 4)) + half * 8;
                bf.u[j] = *(const unsigned*)&sBT[bcol * BSTR + kb];
            }
            acc[ct] = __builtin_amdgcn_wmma_f32_16x16x32_bf16(false, af.v, false, bf.v,
                                                              (short)0, acc[ct], false, false);
        }
        __syncthreads();
    }

    // Epilogue: bias (+ optional fused skip/residual blend).
    float cmul = 1.0f, xmul = 0.0f;
    if (xres) {
        float s = fast_sigmoid(skip[0]);
        cmul = 0.5f * s;                 // (1-R_ALPHA) * s
        xmul = 0.5f * (1.0f - s) + 0.5f; // (1-R_ALPHA)*(1-s) + R_ALPHA
    }
    const int ncol0 = colBase + (lane & 15);
    #pragma unroll
    for (int ct = 0; ct < 8; ++ct) {
        const int ncol = ncol0 + ct * 16;
        const float bv = bias[ncol];
        #pragma unroll
        for (int j = 0; j < 8; ++j) {
            int grow = rowBase + wave * 16 + j + half * 8;
            if (grow < M) {
                float v = acc[ct][j] + bv;
                if (xres) v = cmul * v + xmul * xres[(size_t)grow * HID + ncol];
                C[(size_t)grow * HID + ncol] = v;
            }
        }
    }
}

// ---------------------------------------------------------------------------
// Fold per-head relation transform into weights: W'[c, h*32+e] = sum_d W[c,h*32+d]*a[h,d,e]
// ---------------------------------------------------------------------------
__global__ __launch_bounds__(256)
void hgt_fold_weight(const float* __restrict__ W, const float* __restrict__ A3,
                     float* __restrict__ Wout)
{
    int idx = blockIdx.x * blockDim.x + threadIdx.x;   // 65536
    int c = idx >> 8, j = idx & 255;
    int h = j >> 5, e = j & 31;
    const float* wrow = W + (size_t)c * HID + h * DHEAD;
    const float* arel = A3 + h * DHEAD * DHEAD + e;
    float acc = 0.0f;
    #pragma unroll
    for (int d = 0; d < DHEAD; ++d) acc += wrow[d] * arel[d * DHEAD];
    Wout[idx] = acc;
}

__global__ void hgt_fold_bias(const float* __restrict__ b, const float* __restrict__ A3,
                              float* __restrict__ bout)
{
    int j = threadIdx.x;                                // 256
    int h = j >> 5, e = j & 31;
    float acc = 0.0f;
    #pragma unroll
    for (int d = 0; d < DHEAD; ++d) acc += b[h * DHEAD + d] * A3[h * DHEAD * DHEAD + d * DHEAD + e];
    bout[j] = acc;
}

__global__ void hgt_init_f32(float* __restrict__ p, float val, int n)
{
    int i = blockIdx.x * blockDim.x + threadIdx.x;
    if (i < n) p[i] = val;
}

// ---------------------------------------------------------------------------
// Edge phase: logits + segment max, exp + segment sum, weighted scatter.
// Node rows (q/kr/vr) are L2-resident (192MB L2 > per-relation working set).
// ---------------------------------------------------------------------------
__global__ __launch_bounds__(256)
void hgt_edge_logits(const float* __restrict__ q, const float* __restrict__ kr,
                     const int* __restrict__ src, const int* __restrict__ dst,
                     const float* __restrict__ prel, int E,
                     float* __restrict__ logits, float* __restrict__ mx)
{
    int idx = blockIdx.x * blockDim.x + threadIdx.x;
    if (idx >= E * NHEAD) return;
    int e = idx >> 3, h = idx & 7;
    int s = src[e], d = dst[e];
    const float4* qv = (const float4*)(q  + (size_t)d * HID + h * DHEAD);
    const float4* kv = (const float4*)(kr + (size_t)s * HID + h * DHEAD);
    float acc = 0.0f;
    #pragma unroll
    for (int i = 0; i < 8; ++i) {
        float4 a = qv[i], b = kv[i];
        acc += a.x * b.x + a.y * b.y + a.z * b.z + a.w * b.w;
    }
    float lg = acc * prel[h] * 0.17677669529663687f;   // 1/sqrt(32)
    logits[idx] = lg;
    atomicMax(mx + (size_t)d * NHEAD + h, lg);         // global_atomic_max_num_f32
}

__global__ __launch_bounds__(256)
void hgt_edge_exp(const int* __restrict__ dst, int E,
                  float* __restrict__ logits, const float* __restrict__ mx,
                  float* __restrict__ den)
{
    int idx = blockIdx.x * blockDim.x + threadIdx.x;
    if (idx >= E * NHEAD) return;
    int e = idx >> 3, h = idx & 7;
    int d = dst[e];
    float ex = __builtin_amdgcn_exp2f(1.4426950408889634f * (logits[idx] - mx[(size_t)d * NHEAD + h]));
    logits[idx] = ex;
    atomicAdd(den + (size_t)d * NHEAD + h, ex);
}

__global__ __launch_bounds__(256)
void hgt_edge_aggregate(const float* __restrict__ vr, const int* __restrict__ src,
                        const int* __restrict__ dst, int E,
                        const float* __restrict__ logits, const float* __restrict__ den,
                        float* __restrict__ agg)
{
    int idx = blockIdx.x * blockDim.x + threadIdx.x;
    if (idx >= E * NHEAD) return;
    int e = idx >> 3, h = idx & 7;
    int s = src[e], d = dst[e];
    float alpha = logits[idx] / (den[(size_t)d * NHEAD + h] + 1e-16f);
    const float4* vv = (const float4*)(vr + (size_t)s * HID + h * DHEAD);
    float* ab = agg + (size_t)d * HID + h * DHEAD;
    #pragma unroll
    for (int i = 0; i < 8; ++i) {
        float4 v = vv[i];
        atomicAdd(ab + i * 4 + 0, v.x * alpha);
        atomicAdd(ab + i * 4 + 1, v.y * alpha);
        atomicAdd(ab + i * 4 + 2, v.z * alpha);
        atomicAdd(ab + i * 4 + 3, v.w * alpha);
    }
}

// ---------------------------------------------------------------------------
extern "C" void kernel_launch(void* const* d_in, const int* in_sizes, int n_in,
                              void* d_out, int out_size, void* d_ws, size_t ws_size,
                              hipStream_t stream)
{
    const float* x_user  = (const float*)d_in[0];
    const float* x_item  = (const float*)d_in[1];
    const float* x_taste = (const float*)d_in[2];
    const float* x_image = (const float*)d_in[3];
    const int* eb_src  = (const int*)d_in[4];
    const int* eb_dst  = (const int*)d_in[5];
    const int* eta_src = (const int*)d_in[6];
    const int* eta_dst = (const int*)d_in[7];
    const int* eia_src = (const int*)d_in[8];
    const int* eia_dst = (const int*)d_in[9];
    const float* Wk[3] = { (const float*)d_in[10], (const float*)d_in[14], (const float*)d_in[18] };
    const float* bk[3] = { (const float*)d_in[11], (const float*)d_in[15], (const float*)d_in[19] };
    const float* Wv[3] = { (const float*)d_in[12], (const float*)d_in[16], (const float*)d_in[20] };
    const float* bv[3] = { (const float*)d_in[13], (const float*)d_in[17], (const float*)d_in[21] };
    const float* Wq = (const float*)d_in[22];
    const float* bq = (const float*)d_in[23];
    const float* arel[3] = { (const float*)d_in[24], (const float*)d_in[27], (const float*)d_in[30] };
    const float* mrel[3] = { (const float*)d_in[25], (const float*)d_in[28], (const float*)d_in[31] };
    const float* prel[3] = { (const float*)d_in[26], (const float*)d_in[29], (const float*)d_in[32] };
    const float* Wo   = (const float*)d_in[33];
    const float* bo   = (const float*)d_in[34];
    const float* skip = (const float*)d_in[35];
    float* out = (float*)d_out;

    const float* xsrc[3] = { x_user, x_taste, x_image };
    const int    nsrc[3] = { NU_, NT_, NIM_ };
    const int*   esrc[3] = { eb_src, eta_src, eia_src };
    const int*   edst[3] = { eb_dst, eta_dst, eia_dst };
    const int    necnt[3]= { EB_, ETA_, EIA_ };

    // Workspace bump allocator (256B aligned slabs).
    char* ws = (char*)d_ws;
    auto alloc = [&](size_t bytes) -> float* {
        float* p = (float*)ws;
        ws += (bytes + 255) & ~(size_t)255;
        return p;
    };
    float* q      = alloc((size_t)NI_ * HID * 4);
    float* kr     = alloc((size_t)NT_ * HID * 4);   // max src count rows
    float* vr     = alloc((size_t)NT_ * HID * 4);
    float* agg    = alloc((size_t)NI_ * HID * 4);
    float* logits = alloc((size_t)EB_ * NHEAD * 4); // max edge count
    float* mx     = alloc((size_t)NI_ * NHEAD * 4);
    float* den    = alloc((size_t)NI_ * NHEAD * 4);
    float* Wkp    = alloc((size_t)HID * HID * 4);
    float* bkp    = alloc((size_t)HID * 4);
    float* Wvp    = alloc((size_t)HID * HID * 4);
    float* bvp    = alloc((size_t)HID * 4);
    (void)ws_size; (void)in_sizes; (void)n_in; (void)out_size;

    const dim3 blk(256);
    const dim3 gemmGridNI((NI_ + TM - 1) / TM, HID / TN);

    // q = x_item @ Wq + bq
    hgt_gemm_bf16_wmma<<<gemmGridNI, blk, 0, stream>>>(
        x_item, Wq, bq, q, NI_, 0, nullptr, nullptr);

    // agg = 0
    hgt_init_f32<<<dim3((NI_ * HID + 255) / 256), blk, 0, stream>>>(agg, 0.0f, NI_ * HID);

    for (int r = 0; r < 3; ++r) {
        const int N = nsrc[r], E = necnt[r];

        // Fold relation transforms into projection weights: W' = Wk (x)_h a_rel
        hgt_fold_weight<<<dim3(HID * HID / 256), blk, 0, stream>>>(Wk[r], arel[r], Wkp);
        hgt_fold_bias  <<<dim3(1), blk, 0, stream>>>(bk[r], arel[r], bkp);
        hgt_fold_weight<<<dim3(HID * HID / 256), blk, 0, stream>>>(Wv[r], mrel[r], Wvp);
        hgt_fold_bias  <<<dim3(1), blk, 0, stream>>>(bv[r], mrel[r], bvp);

        // kr = x @ Wk' + bk' ; vr = x @ Wv' + bv'   (fused k->kr, v->vr)
        const dim3 gemmGrid((N + TM - 1) / TM, HID / TN);
        hgt_gemm_bf16_wmma<<<gemmGrid, blk, 0, stream>>>(
            xsrc[r], Wkp, bkp, kr, N, 0, nullptr, nullptr);
        hgt_gemm_bf16_wmma<<<gemmGrid, blk, 0, stream>>>(
            xsrc[r], Wvp, bvp, vr, N, 0, nullptr, nullptr);

        // Segment softmax state
        hgt_init_f32<<<dim3((NI_ * NHEAD + 255) / 256), blk, 0, stream>>>(mx, -3.0e38f, NI_ * NHEAD);
        hgt_init_f32<<<dim3((NI_ * NHEAD + 255) / 256), blk, 0, stream>>>(den, 0.0f, NI_ * NHEAD);

        const int eh = E * NHEAD;
        hgt_edge_logits<<<dim3((eh + 255) / 256), blk, 0, stream>>>(
            q, kr, esrc[r], edst[r], prel[r], E, logits, mx);
        hgt_edge_exp<<<dim3((eh + 255) / 256), blk, 0, stream>>>(
            edst[r], E, logits, mx, den);
        hgt_edge_aggregate<<<dim3((eh + 255) / 256), blk, 0, stream>>>(
            vr, esrc[r], edst[r], E, logits, den, agg);
    }

    // out = blend( sigmoid(skip), gelu(agg) @ Wo + bo, x_item )  -- fully fused
    hgt_gemm_bf16_wmma<<<gemmGridNI, blk, 0, stream>>>(
        agg, Wo, bo, out, NI_, /*preGelu=*/1, x_item, skip);
}
